// SkipGramLoss_88974542504016
// MI455X (gfx1250) — matrix operand order repaired and checked by reading, not compiled, not run on hardware
//
#include <hip/hip_runtime.h>

typedef __attribute__((ext_vector_type(16))) _Float16 v16h;
typedef __attribute__((ext_vector_type(8)))  float    v8f;

#define DIM 128

// Load 8 floats at p and 8 floats at p+second_off, convert to 16 packed halfs.
// All addresses are 32B-aligned (row base is 512B aligned, offsets are
// multiples of 8 floats), so these lower to global_load_b128 pairs.
__device__ __forceinline__ v16h load_frag(const float* __restrict__ p, int second_off) {
  float4 x0 = *reinterpret_cast<const float4*>(p);
  float4 x1 = *reinterpret_cast<const float4*>(p + 4);
  float4 x2 = *reinterpret_cast<const float4*>(p + second_off);
  float4 x3 = *reinterpret_cast<const float4*>(p + second_off + 4);
  v16h f;
  f[0]  = (_Float16)x0.x;  f[1]  = (_Float16)x0.y;
  f[2]  = (_Float16)x0.z;  f[3]  = (_Float16)x0.w;
  f[4]  = (_Float16)x1.x;  f[5]  = (_Float16)x1.y;
  f[6]  = (_Float16)x1.z;  f[7]  = (_Float16)x1.w;
  f[8]  = (_Float16)x2.x;  f[9]  = (_Float16)x2.y;
  f[10] = (_Float16)x2.z;  f[11] = (_Float16)x2.w;
  f[12] = (_Float16)x3.x;  f[13] = (_Float16)x3.y;
  f[14] = (_Float16)x3.z;  f[15] = (_Float16)x3.w;
  return f;
}

// Branch-free variable extract from v8f (cndmask chain, keeps EXEC uniform).
__device__ __forceinline__ float pick8(v8f a, int i) {
  float r = a[0];
#pragma unroll
  for (int j = 1; j < 8; ++j) r = (i == j) ? a[j] : r;
  return r;
}

// Numerically stable softplus: log(1 + e^z)
__device__ __forceinline__ float softplus(float z) {
  return fmaxf(z, 0.0f) + log1pf(__expf(-fabsf(z)));
}

// One wave (32 lanes) per tile of 16 batch elements.
// A = 16 target rows (16x128 f16, 4 K-chunks of 32), resident for the tile.
// For each of the 21 "other" vectors (context + 20 negatives):
//   B = 16 other rows as 128x16 (column n = other vector of batch b0+n)
//   D = A x B via 4x v_wmma_f32_16x16x32_f16; diag(D) = batched dot products.
__global__ __launch_bounds__(32) void skipgram_loss_kernel(
    const float* __restrict__ emb,
    const int*   __restrict__ targets,
    const int*   __restrict__ contexts,
    const int*   __restrict__ negatives,
    float*       __restrict__ out,
    int batch, int num_neg)
{
  const int lane = threadIdx.x;      // 0..31
  const int m    = lane & 15;        // row within tile (both half-waves)
  const int hi   = lane >> 4;        // 0 = lanes 0-15, 1 = lanes 16-31
  const int b0   = blockIdx.x * 16;
  const int b    = b0 + m;

  // ---- A fragment: target rows, ISA 16-bit A layout -----------------------
  // lanes 0-15 hold K = {0..7, 16..23} of each 32-chunk; lanes 16-31 hold
  // K = {8..15, 24..31}  =>  per lane: 8 floats at (chunk + hi*8) and 8 at +16.
  const float* trow = emb + (long)targets[b] * DIM;
  v16h a_frag[4];
#pragma unroll
  for (int c = 0; c < 4; ++c)
    a_frag[c] = load_frag(trow + c * 32 + hi * 8, 16);

  const float invB  = 1.0f / (float)batch;
  const float invBN = invB / (float)num_neg;

  // Diagonal of 16x16 f32 D: lanes 0-7 hold (m=lane, elem=lane),
  // lanes 24-31 hold (m=lane-16, elem=lane-24).
  const bool act0 = (lane < 8);
  const bool act1 = (lane >= 24);
  const int  di   = act0 ? lane : (act1 ? (lane - 24) : 0);

  float lacc = 0.0f;

  for (int chain = 0; chain <= num_neg; ++chain) {
    const long oidx = (chain == 0)
        ? (long)contexts[b]
        : (long)negatives[(long)b * num_neg + (chain - 1)];
    const float* orow = emb + oidx * DIM;

    v8f acc = {};
#pragma unroll
    for (int c = 0; c < 4; ++c) {
      // ---- B fragment: ISA 16-bit B layout (column N = lane%16) ----------
      // lanes 0-15 hold K = {0..7, 16..23} of the chunk, lanes 16-31 K = {8..15, 24..31}:
      // 8 floats at (chunk + hi*8) and 8 at +16 (same order as A).
      v16h b_frag = load_frag(orow + c * 32 + hi * 8, 16);
      acc = __builtin_amdgcn_wmma_f32_16x16x32_f16(
          /*neg_a=*/false, a_frag[c], /*neg_b=*/false, b_frag,
          /*c_mod=*/(short)0, acc, /*reuse_a=*/false, /*reuse_b=*/false);
    }

    const float s = pick8(acc, di);
    // chain 0 (positive): softplus(-s)/B ; negatives: softplus(s)/(B*N)
    const float x = (chain == 0) ? -s : s;
    const float w = (chain == 0) ? invB : invBN;
    lacc += (act0 || act1) ? (w * softplus(x)) : 0.0f;
  }

  // Wave-level reduction, one fp32 atomic per tile (2048 total).
#pragma unroll
  for (int off = 16; off > 0; off >>= 1)
    lacc += __shfl_xor(lacc, off, 32);
  if (lane == 0) atomicAdd(out, lacc);
}

__global__ void zero_out_kernel(float* o) { o[0] = 0.0f; }

extern "C" void kernel_launch(void* const* d_in, const int* in_sizes, int n_in,
                              void* d_out, int out_size, void* d_ws, size_t ws_size,
                              hipStream_t stream) {
  const float* emb       = (const float*)d_in[0];
  const int*   targets   = (const int*)d_in[1];
  const int*   contexts  = (const int*)d_in[2];
  const int*   negatives = (const int*)d_in[3];
  float*       out       = (float*)d_out;

  const int batch   = in_sizes[1];
  const int num_neg = in_sizes[3] / batch;
  const int tiles   = batch / 16;   // batch = 32768 -> 2048 tiles

  zero_out_kernel<<<1, 1, 0, stream>>>(out);
  skipgram_loss_kernel<<<tiles, 32, 0, stream>>>(
      emb, targets, contexts, negatives, out, batch, num_neg);
}
